// LCAU_Block_26233660244750
// MI455X (gfx1250) — matrix operand order repaired and checked by reading, not compiled, not run on hardware
//
#include <hip/hip_runtime.h>

typedef __attribute__((ext_vector_type(16))) _Float16 v16h;
typedef __attribute__((ext_vector_type(8)))  float    v8f;

#define BATCH 4
#define CIN   128
#define HIN   64
#define WIN   64
#define L2CH  25      // 5x5 kernel weights
#define OUTC  128
#define NCHUNK 4      // 128 channels / 32
#define CCH    32     // channels per chunk
#define XCOLS  70     // 68 used (W + 4 halo); 70 => 8B-aligned b64 async + conflict-free strided reads

// ---------------- prep: pack w1 into WMMA A-fragment layout (f16) ----------------
// layout: a1[chunk][shift][mtile][lane][16 halves]; total 4*9*2*32*16 = 36864 halves
__global__ void prep_w1_frag(const float* __restrict__ w1, _Float16* __restrict__ a1) {
    int idx = blockIdx.x * 256 + threadIdx.x;
    if (idx >= 4 * 9 * 2 * 32 * 16) return;
    int e     = idx & 15;
    int lane  = (idx >> 4) & 31;
    int mtile = (idx >> 9) & 1;
    int shift = (idx >> 10) % 9;
    int chunk = idx / 9216;
    // A 16x32 f16 layout (ISA 7.12.2): v<4: K = 8*hi + 2v + pos ; v>=4: K = 16 + 8*hi + 2(v-4) + pos
    int v = e >> 1, pos = e & 1, hi = lane >> 4;
    int K = (v < 4) ? (8 * hi + 2 * v + pos) : (16 + 8 * hi + 2 * (v - 4) + pos);
    int row = mtile * 16 + (lane & 15);          // conv1 output channel (0..24 valid)
    int dy = shift / 3, dx = shift % 3;
    float val = 0.f;
    if (row < L2CH) {
        int c = chunk * CCH + K;
        val = w1[((row * CIN + c) * 3 + dy) * 3 + dx];
    }
    a1[idx] = (_Float16)val;
}

// ---------------- prep: fold w2 over r (4C->C) and pack A-fragments ----------------
// layout: a2[mtile8][chunk4][lane][16 halves]; total 8*4*32*16 = 16384 halves
__global__ void prep_w2_frag(const float* __restrict__ w2, _Float16* __restrict__ a2) {
    int idx = blockIdx.x * 256 + threadIdx.x;
    if (idx >= 8 * 4 * 32 * 16) return;
    int e     = idx & 15;
    int lane  = (idx >> 4) & 31;
    int chunk = (idx >> 9) & 3;
    int mtile = idx >> 11;
    int v = e >> 1, pos = e & 1, hi = lane >> 4;
    int K = (v < 4) ? (8 * hi + 2 * v + pos) : (16 + 8 * hi + 2 * (v - 4) + pos);
    int row = mtile * 16 + (lane & 15);
    int c = chunk * CCH + K;
    const float* p = w2 + (size_t)row * (4 * CIN) + c * 4;
    // nearest-upsample + softmax identity: all 4 subpixel kernels equal ->
    // fold w2 over the R^2 dim exactly.
    float val = p[0] + p[1] + p[2] + p[3];
    a2[idx] = (_Float16)val;
}

// -------- async staging: x[b, chunk*32..+31, h-2..h+2, 0..63] -> LDS (halo pre-zeroed) --------
__device__ __forceinline__ void stage_x_async(float (*xs)[CCH][XCOLS], const float* __restrict__ xb,
                                              int chunk, int h, int t) {
    const int q = t & 31;                                   // lane: covers gw = 2q, 2q+1
    const unsigned long long gbase =
        (unsigned long long)(uintptr_t)(xb + (size_t)chunk * CCH * HIN * WIN);
#pragma unroll
    for (int i = 0; i < 20; ++i) {
        int rc  = (t >> 5) + 8 * i;                         // wave-uniform (row, cc) slice
        int row = rc >> 5, cc = rc & 31;
        int gh  = h - 2 + row;
        if ((unsigned)gh < (unsigned)HIN) {
            unsigned voff = (unsigned)(((cc * HIN + gh) * WIN + 2 * q) * 4);
            unsigned lds  = (unsigned)(uintptr_t)&xs[row][cc][2 + 2 * q];   // 8B aligned
            asm volatile("global_load_async_to_lds_b64 %0, %1, %2"
                         :: "v"(lds), "v"(voff), "s"(gbase) : "memory");
        }
    }
    asm volatile("s_wait_asynccnt 0" ::: "memory");
}

__global__ __launch_bounds__(256)
void lcau_fused(const float* __restrict__ x,  const float* __restrict__ b1,
                const float* __restrict__ b2, const _Float16* __restrict__ a1,
                const _Float16* __restrict__ a2, float* __restrict__ out) {
    __shared__ float xs[5][CCH][XCOLS];                       // 44800 B
    __shared__ float ks[L2CH][WIN];                           //  6400 B (k, then softmax s)
    __shared__ __attribute__((aligned(32))) _Float16 fB[4][32][16]; // 4096 B (B fragments)

    const int t    = threadIdx.x;
    const int lane = t & 31;
    const int wave = t >> 5;
    const int b    = blockIdx.x >> 6;
    const int h    = blockIdx.x & 63;
    const float* xb = x + (size_t)b * CIN * HIN * WIN;

    const int n  = lane & 15;
    const int hi = lane >> 4;

    // ---- one-time zero of halo columns + out-of-range rows (persists across restages) ----
    for (int u = t; u < 5 * CCH * 4; u += 256) {
        int cidx = u & 3;                                   // -> cols 0,1,66,67
        int rc = u >> 2;
        xs[rc >> 5][rc & 31][(cidx < 2) ? cidx : 64 + cidx] = 0.f;
    }
    for (int row = 0; row < 5; ++row) {
        int gh = h - 2 + row;
        if (gh < 0 || gh >= HIN)
            for (int u = t; u < CCH * 68; u += 256)
                xs[row][u / 68][u % 68] = 0.f;
    }

    // ================= Phase 1: conv1 (3x3) via WMMA =================
    const int mt1 = wave >> 2;   // 0..1  (rows 0-15 / 16-31, valid rows < 25)
    const int nt1 = wave & 3;    // 0..3  (pixel tiles of 16)
    v8f acc1 = {};
    for (int chunk = 0; chunk < NCHUNK; ++chunk) {
        stage_x_async(xs, xb, chunk, h, t);
        __syncthreads();
        for (int shift = 0; shift < 9; ++shift) {
            int dyr = shift / 3 + 1;                 // xs row for h + dy0 - 1
            int dx  = shift % 3;
            int col = nt1 * 16 + n + dx + 1;         // xs col for w + dx0 - 1
            // B 32x16 f16 fragment: lanes 0-15 hold K=0-7,16-23, lanes 16-31 hold K=8-15,24-31
            v16h bfrag;
#pragma unroll
            for (int e = 0; e < 16; ++e)
                bfrag[e] = (_Float16)xs[dyr][8 * hi + e + (e & 8)][col];
            v16h afrag = *(const v16h*)(a1 + ((((size_t)chunk * 9 + shift) * 2 + mt1) * 32 + lane) * 16);
            acc1 = __builtin_amdgcn_wmma_f32_16x16x32_f16(false, afrag, false, bfrag,
                                                          (short)0, acc1, false, false);
        }
        __syncthreads();
    }
    {   // write k tile (+bias) to LDS; C/D layout: vgpr r -> M = r + 8*hi, N = n
        int p = nt1 * 16 + n;
#pragma unroll
        for (int r = 0; r < 8; ++r) {
            int m = mt1 * 16 + hi * 8 + r;
            if (m < L2CH) ks[m][p] = acc1[r] + b1[m];
        }
    }
    __syncthreads();

    // ================= Phase 2: softmax over 25 (per pixel) =================
    if (t < WIN) {
        float kv[L2CH];
        float mx = -3.4e38f;
#pragma unroll
        for (int j = 0; j < L2CH; ++j) { kv[j] = ks[j][t]; mx = fmaxf(mx, kv[j]); }
        float sum = 0.f;
#pragma unroll
        for (int j = 0; j < L2CH; ++j) { kv[j] = __expf(kv[j] - mx); sum += kv[j]; }
        float inv = 1.f / sum;
#pragma unroll
        for (int j = 0; j < L2CH; ++j) ks[j][t] = kv[j] * inv;
    }
    __syncthreads();

    // ============ Phase 3+4: 5x5 per-pixel filtering + 1x1 GEMM via WMMA ============
    v8f acc2[4] = {v8f{}, v8f{}, v8f{}, v8f{}};   // wave = mtile of OUT (8 x 16 rows)
    const int K    = lane;                        // this thread's channel within chunk
    const int khi  = (K >> 3) & 1;
    const int kelm = (K & 7) + ((K >> 4) << 3);
    for (int chunk = 0; chunk < NCHUNK; ++chunk) {
        stage_x_async(xs, xb, chunk, h, t);
        __syncthreads();
        // filter: f[c,p] = sum_{dy,dx} s[5dy+dx, p] * x[c, h+dy-2, p+dx-2]
#pragma unroll
        for (int j = 0; j < 8; ++j) {
            int p = wave + 8 * j;                 // wave-uniform pixel -> s broadcast
            float f = 0.f;
#pragma unroll
            for (int dy = 0; dy < 5; ++dy)
#pragma unroll
                for (int dx = 0; dx < 5; ++dx)
                    f += ks[dy * 5 + dx][p] * xs[dy][lane][p + dx];
            fB[p >> 4][(p & 15) + 16 * khi][kelm] = (_Float16)f;  // B-fragment slot
        }
        __syncthreads();
        // GEMM: out_tile[wave] += W2eff[:, chunk] @ f[chunk]
        v16h afrag = *(const v16h*)(a2 + (((size_t)wave * 4 + chunk) * 32 + lane) * 16);
#pragma unroll
        for (int nt = 0; nt < 4; ++nt) {
            v16h bfrag = *(const v16h*)&fB[nt][lane][0];
            acc2[nt] = __builtin_amdgcn_wmma_f32_16x16x32_f16(false, afrag, false, bfrag,
                                                              (short)0, acc2[nt], false, false);
        }
        __syncthreads();
    }

    // ============ epilogue: bias + exact 2x2 nearest replication to [4,128,128,128] ============
#pragma unroll
    for (int nt = 0; nt < 4; ++nt) {
        int p = nt * 16 + n;
#pragma unroll
        for (int r = 0; r < 8; ++r) {
            int m = wave * 16 + hi * 8 + r;
            float vo = acc2[nt][r] + b2[m];
            size_t base = (((size_t)b * OUTC + m) * (2 * HIN) + 2 * h) * (2 * WIN) + 2 * p;
            float2 vv = make_float2(vo, vo);
            *(float2*)(out + base)            = vv;   // row 2h,   cols 2p,2p+1
            *(float2*)(out + base + 2 * WIN)  = vv;   // row 2h+1, cols 2p,2p+1
        }
    }
}

extern "C" void kernel_launch(void* const* d_in, const int* in_sizes, int n_in,
                              void* d_out, int out_size, void* d_ws, size_t ws_size,
                              hipStream_t stream) {
    (void)in_sizes; (void)n_in; (void)out_size; (void)ws_size;
    const float* x  = (const float*)d_in[0];
    const float* w1 = (const float*)d_in[1];
    const float* b1 = (const float*)d_in[2];
    const float* w2 = (const float*)d_in[3];
    const float* b2 = (const float*)d_in[4];
    float* out = (float*)d_out;

    _Float16* a1 = (_Float16*)d_ws;               // 36864 halves = 73728 B
    _Float16* a2 = a1 + 4 * 9 * 2 * 32 * 16;      // 16384 halves = 32768 B

    prep_w1_frag<<<(36864 + 255) / 256, 256, 0, stream>>>(w1, a1);
    prep_w2_frag<<<(16384 + 255) / 256, 256, 0, stream>>>(w2, a2);
    lcau_fused<<<BATCH * HIN, 256, 0, stream>>>(x, b1, b2, a1, a2, out);
}
